// EnhancedSwitchMLP_59863254171938
// MI455X (gfx1250) — matrix operand order repaired and checked by reading, not compiled, not run on hardware
//
#include <hip/hip_runtime.h>
#include <hip/hip_bf16.h>

// ---- problem dims (from reference) ----
#define T_TOK 4096   // B*S = 2*2048
#define H_DIM 1024
#define E_NUM 8
#define I_DIM 2688
#define MAXK  6
#define AHID  50

// ---- GEMM tiling ----
#define BM 64
#define BN 128
#define BK 32        // one v_wmma_f32_16x16x32_bf16 K-step

typedef __attribute__((ext_vector_type(16))) __bf16 v16bf;
typedef __attribute__((ext_vector_type(8)))  float  v8f;

union FragBF {
    v16bf v;
    uint4 q[2];
};

// ---- CDNA5 async copy: global -> LDS, 16B per lane, ASYNCcnt tracked ----
__device__ __forceinline__ void async_load_b128(unsigned lds_off, const void* gptr)
{
    unsigned long long ga = (unsigned long long)(size_t)gptr;
    asm volatile("global_load_async_to_lds_b128 %0, %1, off"
                 :: "v"(lds_off), "v"(ga)
                 : "memory");
}
__device__ __forceinline__ void wait_async()
{
    asm volatile("s_wait_asynccnt 0x0" ::: "memory");
}

// pack two f32 -> one dword of two bf16
__device__ __forceinline__ unsigned pk_bf16(float lo, float hi)
{
    union { __hip_bfloat16 h; unsigned short u; } a, b;
    a.h = __float2bfloat16(lo);
    b.h = __float2bfloat16(hi);
    return ((unsigned)b.u << 16) | (unsigned)a.u;
}

// 16 f32 (global) -> 16 bf16 (LDS) as two ds_store_b128
__device__ __forceinline__ void stage_w16(__hip_bfloat16* dst, const float* src)
{
    const float4* s4 = reinterpret_cast<const float4*>(src);
    const float4 f0 = s4[0], f1 = s4[1], f2 = s4[2], f3 = s4[3];
    uint4 lo, hi;
    lo.x = pk_bf16(f0.x, f0.y); lo.y = pk_bf16(f0.z, f0.w);
    lo.z = pk_bf16(f1.x, f1.y); lo.w = pk_bf16(f1.z, f1.w);
    hi.x = pk_bf16(f2.x, f2.y); hi.y = pk_bf16(f2.z, f2.w);
    hi.z = pk_bf16(f3.x, f3.y); hi.w = pk_bf16(f3.z, f3.w);
    uint4* o = reinterpret_cast<uint4*>(dst);
    o[0] = lo;
    o[1] = hi;
}

// =====================================================================
// Phase A: router softmax + actor MLP -> per-token top-k scores [T,E]
// One wave (32 lanes) per token; butterfly reductions via __shfl_xor.
// =====================================================================
__global__ __launch_bounds__(256)
void router_actor_kernel(const float* __restrict__ x,    // [T,H]
                         const float* __restrict__ rw,   // [E,H]
                         const float* __restrict__ w1,   // [AHID,H]
                         const float* __restrict__ b1,   // [AHID]
                         const float* __restrict__ w2,   // [MAXK,AHID]
                         const float* __restrict__ b2,   // [MAXK]
                         float* __restrict__ scores)     // [T,E]
{
    const int wave = (blockIdx.x * blockDim.x + threadIdx.x) >> 5;
    const int lane = threadIdx.x & 31;
    if (wave >= T_TOK) return;

    const float* xt = x + (size_t)wave * H_DIM;
    float xr[32];
#pragma unroll
    for (int j = 0; j < 32; j++) xr[j] = xt[lane + 32 * j];

    // ---- router logits + softmax ----
    float prob[E_NUM];
#pragma unroll
    for (int e = 0; e < E_NUM; e++) {
        const float* w = rw + (size_t)e * H_DIM;
        float p = 0.f;
#pragma unroll
        for (int j = 0; j < 32; j++) p = fmaf(xr[j], w[lane + 32 * j], p);
#pragma unroll
        for (int m = 16; m > 0; m >>= 1) p += __shfl_xor(p, m, 32);
        prob[e] = p;
    }
    float mx = prob[0];
#pragma unroll
    for (int e = 1; e < E_NUM; e++) mx = fmaxf(mx, prob[e]);
    float sum = 0.f;
#pragma unroll
    for (int e = 0; e < E_NUM; e++) { prob[e] = __expf(prob[e] - mx); sum += prob[e]; }
    const float inv = 1.f / sum;
#pragma unroll
    for (int e = 0; e < E_NUM; e++) prob[e] *= inv;

    // ---- actor: Linear -> GELU(tanh) -> Linear -> clip -> argmax+1 ----
    float al[MAXK];
#pragma unroll
    for (int k = 0; k < MAXK; k++) al[k] = b2[k];
    for (int a = 0; a < AHID; a++) {
        const float* w = w1 + (size_t)a * H_DIM;
        float p = 0.f;
#pragma unroll
        for (int j = 0; j < 32; j++) p = fmaf(xr[j], w[lane + 32 * j], p);
#pragma unroll
        for (int m = 16; m > 0; m >>= 1) p += __shfl_xor(p, m, 32);
        p += b1[a];
        const float c = p + 0.044715f * p * p * p;
        const float g = 0.5f * p * (1.f + tanhf(0.7978845608028654f * c));
#pragma unroll
        for (int k = 0; k < MAXK; k++) al[k] += g * w2[k * AHID + a];
    }
    int   best = 0;
    float bv   = fminf(fmaxf(al[0], -30.f), 30.f);
#pragma unroll
    for (int k = 1; k < MAXK; k++) {
        const float v = fminf(fmaxf(al[k], -30.f), 30.f);
        if (v > bv) { bv = v; best = k; }
    }
    const int kk = best + 1;

    // ---- rank-based top-k mask (stable like argsort) ----
    if (lane < E_NUM) {
        const float pe = prob[lane];
        int rank = 0;
#pragma unroll
        for (int e2 = 0; e2 < E_NUM; e2++) {
            const float po = prob[e2];
            rank += (po > pe) || (po == pe && e2 < lane);
        }
        scores[(size_t)wave * E_NUM + lane] = (rank < kk) ? pe : 0.f;
    }
}

// =====================================================================
// Phase B: X fp32 -> bf16, and zero the output accumulator
// =====================================================================
__global__ void zero_convert_kernel(const float* __restrict__ x,
                                    __hip_bfloat16* __restrict__ xb,
                                    float* __restrict__ out)
{
    const int n = T_TOK * H_DIM;
    for (int i = blockIdx.x * blockDim.x + threadIdx.x; i < n;
         i += gridDim.x * blockDim.x) {
        xb[i]  = __float2bfloat16(x[i]);
        out[i] = 0.f;
    }
}

// ---- fragment loaders (LDS tile is row-major [rows][BK] bf16) ----
__device__ __forceinline__ void load_frag_a(const __hip_bfloat16* tile, int row,
                                            int hl, FragBF& f)
{
    // A fragment, this code's contiguous k order (B matches): half hl holds K = hl*16..+15
    const __hip_bfloat16* base = tile + row * BK;
    f.q[0] = *reinterpret_cast<const uint4*>(base + hl * 16);
    f.q[1] = *reinterpret_cast<const uint4*>(base + hl * 16 + 8);
}

__device__ __forceinline__ void load_frag_b(const __hip_bfloat16* tile, int row,
                                            int hl, FragBF& f)
{
    // B fragment, same contiguous k order as A (col-major rows = N): half hl holds K = hl*16..+15
    const uint4* p = reinterpret_cast<const uint4*>(tile + row * BK + hl * 16);
    f.q[0] = p[0];
    f.q[1] = p[1];
}

// =====================================================================
// Phase C (per expert): act[t,i] = silu(X.Wg^T) * (X.Wu^T), bf16 out
// BM=64 x BN=128, 8 waves as 2x4, wave tile 32x32, fused gate+up.
// Ping-pong LDS; A-tile via global_load_async_to_lds_b128 (ASYNCcnt).
// =====================================================================
__global__ __launch_bounds__(256)
void gateup_kernel(const __hip_bfloat16* __restrict__ Xb, // [T,H] bf16
                   const float* __restrict__ Wg,          // [I,H] fp32 (expert slice)
                   const float* __restrict__ Wu,          // [I,H] fp32 (expert slice)
                   __hip_bfloat16* __restrict__ act)      // [T,I] bf16
{
    __shared__ __hip_bfloat16 AsBuf[2][BM * BK];
    __shared__ __hip_bfloat16 GsBuf[2][BN * BK];
    __shared__ __hip_bfloat16 UsBuf[2][BN * BK];

    const int tid  = threadIdx.x;
    const int lane = tid & 31;
    const int wave = tid >> 5;
    const int wm   = wave & 1;   // 0..1 -> M
    const int wn   = wave >> 1;  // 0..3 -> N
    const int hl   = lane >> 4;  // half-wave
    const int r    = lane & 15;

    const int n0 = blockIdx.x * BN;
    const int m0 = blockIdx.y * BM;

    v8f accG[2][2], accU[2][2];
    const v8f vz = {};
#pragma unroll
    for (int i = 0; i < 2; i++)
#pragma unroll
        for (int j = 0; j < 2; j++) { accG[i][j] = vz; accU[i][j] = vz; }

    // cooperative load mapping
    const int arow = tid >> 2;          // 0..63
    const int acol = (tid & 3) * 8;     // 8 bf16 = one 16B chunk
    const int brow = tid >> 1;          // 0..127
    const int bcol = (tid & 1) * 16;    // 16 floats -> 16 bf16

    const unsigned ldsA[2] = {
        (unsigned)(size_t)&AsBuf[0][arow * BK + acol],
        (unsigned)(size_t)&AsBuf[1][arow * BK + acol]
    };
    const __hip_bfloat16* asrc = &Xb[(size_t)(m0 + arow) * H_DIM + acol];
    const float* gsrc = &Wg[(size_t)(n0 + brow) * H_DIM + bcol];
    const float* usrc = &Wu[(size_t)(n0 + brow) * H_DIM + bcol];

    // ---- prologue: stage k0 = 0 into buffer 0 ----
    async_load_b128(ldsA[0], asrc);
    stage_w16(&GsBuf[0][brow * BK + bcol], gsrc);
    stage_w16(&UsBuf[0][brow * BK + bcol], usrc);
    wait_async();
    __syncthreads();

    int cur = 0;
    for (int k0 = 0; k0 < H_DIM; k0 += BK) {
        const int  nxt      = cur ^ 1;
        const bool has_next = (k0 + BK) < H_DIM;

        // prefetch next K-slab into the other buffer (async A + packed W)
        if (has_next)
            async_load_b128(ldsA[nxt], asrc + (k0 + BK));

        FragBF a[2], bg[2], bu[2];
        const __hip_bfloat16* As = AsBuf[cur];
        const __hip_bfloat16* Gs = GsBuf[cur];
        const __hip_bfloat16* Us = UsBuf[cur];
#pragma unroll
        for (int i = 0; i < 2; i++) load_frag_a(As, wm * 32 + i * 16 + r, hl, a[i]);
#pragma unroll
        for (int j = 0; j < 2; j++) {
            load_frag_b(Gs, wn * 32 + j * 16 + r, hl, bg[j]);
            load_frag_b(Us, wn * 32 + j * 16 + r, hl, bu[j]);
        }

#pragma unroll
        for (int i = 0; i < 2; i++)
#pragma unroll
            for (int j = 0; j < 2; j++) {
                accG[i][j] = __builtin_amdgcn_wmma_f32_16x16x32_bf16(
                    false, a[i].v, false, bg[j].v, (short)0, accG[i][j], false, false);
                accU[i][j] = __builtin_amdgcn_wmma_f32_16x16x32_bf16(
                    false, a[i].v, false, bu[j].v, (short)0, accU[i][j], false, false);
            }

        if (has_next) {
            stage_w16(&GsBuf[nxt][brow * BK + bcol], gsrc + (k0 + BK));
            stage_w16(&UsBuf[nxt][brow * BK + bcol], usrc + (k0 + BK));
        }

        wait_async();
        __syncthreads();
        cur = nxt;
    }

    // epilogue: silu(g) * u -> bf16
#pragma unroll
    for (int i = 0; i < 2; i++)
#pragma unroll
        for (int j = 0; j < 2; j++)
#pragma unroll
            for (int d = 0; d < 8; d++) {
                const int t = m0 + wm * 32 + i * 16 + hl * 8 + d;
                const int c = n0 + wn * 32 + j * 16 + r;
                const float g = accG[i][j][d];
                const float u = accU[i][j][d];
                const float s = g / (1.f + __expf(-g));
                act[(size_t)t * I_DIM + c] = __float2bfloat16(s * u);
            }
}

// =====================================================================
// Phase D (per expert): out[t,h] += scores[t,e] * (act . Wd^T)[t,h]
// =====================================================================
__global__ __launch_bounds__(256)
void down_kernel(const __hip_bfloat16* __restrict__ act, // [T,I] bf16
                 const float* __restrict__ Wd,           // [H,I] fp32 (expert slice)
                 const float* __restrict__ scores,       // [T,E]
                 int e,
                 float* __restrict__ out)                // [T,H] accumulate
{
    __shared__ __hip_bfloat16 AsBuf[2][BM * BK];
    __shared__ __hip_bfloat16 BsBuf[2][BN * BK];

    const int tid  = threadIdx.x;
    const int lane = tid & 31;
    const int wave = tid >> 5;
    const int wm   = wave & 1;
    const int wn   = wave >> 1;
    const int hl   = lane >> 4;
    const int r    = lane & 15;

    const int n0 = blockIdx.x * BN;   // H
    const int m0 = blockIdx.y * BM;   // T

    v8f acc[2][2];
    const v8f vz = {};
#pragma unroll
    for (int i = 0; i < 2; i++)
#pragma unroll
        for (int j = 0; j < 2; j++) acc[i][j] = vz;

    const int arow = tid >> 2;
    const int acol = (tid & 3) * 8;
    const int brow = tid >> 1;
    const int bcol = (tid & 1) * 16;

    const unsigned ldsA[2] = {
        (unsigned)(size_t)&AsBuf[0][arow * BK + acol],
        (unsigned)(size_t)&AsBuf[1][arow * BK + acol]
    };
    const __hip_bfloat16* asrc = &act[(size_t)(m0 + arow) * I_DIM + acol];
    const float* dsrc = &Wd[(size_t)(n0 + brow) * I_DIM + bcol];

    // ---- prologue ----
    async_load_b128(ldsA[0], asrc);
    stage_w16(&BsBuf[0][brow * BK + bcol], dsrc);
    wait_async();
    __syncthreads();

    int cur = 0;
    for (int k0 = 0; k0 < I_DIM; k0 += BK) {
        const int  nxt      = cur ^ 1;
        const bool has_next = (k0 + BK) < I_DIM;

        if (has_next)
            async_load_b128(ldsA[nxt], asrc + (k0 + BK));

        FragBF a[2], b[2];
        const __hip_bfloat16* As = AsBuf[cur];
        const __hip_bfloat16* Bs = BsBuf[cur];
#pragma unroll
        for (int i = 0; i < 2; i++) load_frag_a(As, wm * 32 + i * 16 + r, hl, a[i]);
#pragma unroll
        for (int j = 0; j < 2; j++) load_frag_b(Bs, wn * 32 + j * 16 + r, hl, b[j]);

#pragma unroll
        for (int i = 0; i < 2; i++)
#pragma unroll
            for (int j = 0; j < 2; j++)
                acc[i][j] = __builtin_amdgcn_wmma_f32_16x16x32_bf16(
                    false, a[i].v, false, b[j].v, (short)0, acc[i][j], false, false);

        if (has_next)
            stage_w16(&BsBuf[nxt][brow * BK + bcol], dsrc + (k0 + BK));

        wait_async();
        __syncthreads();
        cur = nxt;
    }

#pragma unroll
    for (int i = 0; i < 2; i++)
#pragma unroll
        for (int d = 0; d < 8; d++) {
            const int t  = m0 + wm * 32 + i * 16 + hl * 8 + d;
            const float sc = scores[(size_t)t * E_NUM + e];
#pragma unroll
            for (int j = 0; j < 2; j++) {
                const int h = n0 + wn * 32 + j * 16 + r;
                const size_t idx = (size_t)t * H_DIM + h;
                out[idx] = out[idx] + sc * acc[i][j][d];
            }
        }
}

// =====================================================================
extern "C" void kernel_launch(void* const* d_in, const int* in_sizes, int n_in,
                              void* d_out, int out_size, void* d_ws, size_t ws_size,
                              hipStream_t stream)
{
    const float* x    = (const float*)d_in[0]; // hidden_states [B,S,H]
    const float* rw   = (const float*)d_in[1]; // router_w [E,H]
    const float* w1   = (const float*)d_in[2]; // actor_w1 [AHID,H]
    const float* b1   = (const float*)d_in[3];
    const float* w2   = (const float*)d_in[4]; // actor_w2 [MAXK,AHID]
    const float* b2   = (const float*)d_in[5];
    const float* gate = (const float*)d_in[6]; // [E,I,H]
    const float* up   = (const float*)d_in[7]; // [E,I,H]
    const float* down = (const float*)d_in[8]; // [E,H,I]
    float* out = (float*)d_out;

    // workspace layout (~30.6 MB): scores | X_bf16 | act_bf16
    char* ws = (char*)d_ws;
    float* scores = (float*)ws;
    const size_t scores_bytes = (size_t)T_TOK * E_NUM * sizeof(float);   // 128 KB
    __hip_bfloat16* Xb  = (__hip_bfloat16*)(ws + ((scores_bytes + 255) & ~(size_t)255));
    const size_t xb_bytes = (size_t)T_TOK * H_DIM * sizeof(__hip_bfloat16); // 8 MB
    __hip_bfloat16* actb = (__hip_bfloat16*)((char*)Xb + ((xb_bytes + 255) & ~(size_t)255));

    router_actor_kernel<<<T_TOK / 8, 256, 0, stream>>>(x, rw, w1, b1, w2, b2, scores);
    zero_convert_kernel<<<1024, 256, 0, stream>>>(x, Xb, out);

    for (int e = 0; e < E_NUM; e++) {
        const size_t wgu_off = (size_t)e * I_DIM * H_DIM;
        const size_t wd_off  = (size_t)e * H_DIM * I_DIM;
        gateup_kernel<<<dim3(I_DIM / BN, T_TOK / BM), 256, 0, stream>>>(
            Xb, gate + wgu_off, up + wgu_off, actb);
        down_kernel<<<dim3(H_DIM / BN, T_TOK / BM), 256, 0, stream>>>(
            actb, down + wd_off, scores, e, out);
    }
}
